// SmolDiTAttention_20667382629197
// MI455X (gfx1250) — compile-verified, matrix operand order checked by reading, not run on hardware
//
#include <hip/hip_runtime.h>
#include <hip/hip_bf16.h>

typedef __attribute__((ext_vector_type(16))) __bf16 v16bf;
typedef __attribute__((ext_vector_type(8)))  __bf16 v8bf;
typedef __attribute__((ext_vector_type(4)))  __bf16 v4bf;
typedef __attribute__((ext_vector_type(8)))  float  v8f;
typedef __attribute__((ext_vector_type(4)))  float  v4f;

__device__ __forceinline__ v8f wmma_bf16(v16bf a, v16bf b, v8f c) {
  // D = A(16x32 bf16) x B(32x16 bf16) + C(16x16 f32)
  return __builtin_amdgcn_wmma_f32_16x16x32_bf16(false, a, false, b, (short)0, c,
                                                 false, false);
}

// Load a 16x32 bf16 fragment (A-layout; identical per-lane pattern serves the
// B operand when the tile is stored [n][k] in LDS).
__device__ __forceinline__ v16bf frag_ld(const __bf16* base, int stride, int row16) {
  int lane = threadIdx.x & 31;
  const __bf16* p = base + (size_t)(row16 + (lane & 15)) * stride + ((lane >> 4) << 3);
  v8bf lo = *(const v8bf*)p;
  v8bf hi = *(const v8bf*)(p + 16);
  v16bf f;
#pragma unroll
  for (int i = 0; i < 8; ++i) { f[i] = lo[i]; f[i + 8] = hi[i]; }
  return f;
}

// ---------------------------------------------------------------------------
// Projection GEMM (fp32 in -> bf16 out) with fused RoPE.
// mode 0: Q (24 heads, rope, dst[b][h][s][d])
// mode 1: K ( 8 heads, rope, dst[b][kv][s][d])
// mode 2: V ( 8 heads, no rope, TRANSPOSED dst[b][kv][d][s])
// ---------------------------------------------------------------------------
#define PROJ_PAD 40   // 32 bf16 + 8 pad -> conflict-free ds_load_b128 frags

__global__ __launch_bounds__(256)
void proj_rope_kernel(const float* __restrict__ X, const float* __restrict__ W,
                      const float* __restrict__ emb, __bf16* __restrict__ dst,
                      int S, int QD, int N, int mode) {
  __shared__ __bf16 As[128 * PROJ_PAD];
  __shared__ __bf16 Bs[128 * PROJ_PAD];
  const int tid = threadIdx.x;
  const int lane = tid & 31;
  const int wave = tid >> 5;
  const int wm = wave >> 1, wn = wave & 1;          // 4x2 wave grid
  const int m0 = blockIdx.y * 128, n0 = blockIdx.x * 128;

  v8f acc[2][4];
#pragma unroll
  for (int a = 0; a < 2; ++a)
#pragma unroll
    for (int t = 0; t < 4; ++t)
#pragma unroll
      for (int i = 0; i < 8; ++i) acc[a][t][i] = 0.0f;

  for (int k0 = 0; k0 < QD; k0 += 32) {
    __syncthreads();
    // ---- stage A tile 128x32 (fp32 -> bf16), row-major [m][k]
#pragma unroll
    for (int p = 0; p < 4; ++p) {
      int row = p * 32 + (tid >> 3);
      int c4  = (tid & 7) * 4;
      v4f v = *(const v4f*)&X[(size_t)(m0 + row) * QD + k0 + c4];
      v4bf o;
      o[0] = (__bf16)v[0]; o[1] = (__bf16)v[1];
      o[2] = (__bf16)v[2]; o[3] = (__bf16)v[3];
      *(v4bf*)&As[row * PROJ_PAD + c4] = o;
    }
    // ---- stage B tile 32x128 transposed into [n][k]
#pragma unroll
    for (int p = 0; p < 4; ++p) {
      int k  = p * 8 + (tid >> 5);
      int n4 = (tid & 31) * 4;
      v4f v = *(const v4f*)&W[(size_t)(k0 + k) * N + n0 + n4];
#pragma unroll
      for (int i = 0; i < 4; ++i) Bs[(n4 + i) * PROJ_PAD + k] = (__bf16)v[i];
    }
    __syncthreads();

    v16bf af[2];
#pragma unroll
    for (int mt = 0; mt < 2; ++mt) af[mt] = frag_ld(As, PROJ_PAD, wm * 32 + mt * 16);
#pragma unroll
    for (int nt = 0; nt < 4; ++nt) {
      v16bf bf = frag_ld(Bs, PROJ_PAD, wn * 64 + nt * 16);
#pragma unroll
      for (int mt = 0; mt < 2; ++mt) acc[mt][nt] = wmma_bf16(af[mt], bf, acc[mt][nt]);
    }
  }

  // ---- epilogue: RoPE (modes 0/1) + bf16 store
  // S is a multiple of the 128-row tile, so the whole block is one batch:
  // hoist the division out of the per-element loop (one scalar div per block).
  const int heads = (mode == 0) ? 24 : 8;
  const int bb    = m0 / S;
  const int sb    = m0 - bb * S;      // sequence position of block row 0
#pragma unroll
  for (int mt = 0; mt < 2; ++mt) {
#pragma unroll
    for (int nt = 0; nt < 4; ++nt) {
      int n = n0 + wn * 64 + nt * 16 + (lane & 15);
      int h = n >> 6, d = n & 63, j = d >> 1;
      __bf16* dq = dst + ((size_t)(bb * heads + h) * S) * 64 + d;   // modes 0/1
      __bf16* dv = dst + ((size_t)(bb * 8 + h) * 64 + d) * S;       // mode 2
#pragma unroll
      for (int r = 0; r < 8; ++r) {
        int s = sb + wm * 32 + mt * 16 + r + ((lane >> 4) << 3);
        float v  = acc[mt][nt][r];
        float pp = __shfl_xor(v, 1, 32);   // partner head-dim (adjacent lane)
        if (mode == 2) {
          dv[s] = (__bf16)v;
        } else {
          float cs = emb[((size_t)s * 32 + j) * 2];
          float sn = emb[((size_t)s * 32 + j) * 2 + 1];
          float o  = (d & 1) ? (pp * sn + v * cs) : (v * cs - pp * sn);
          dq[(size_t)s * 64] = (__bf16)o;
        }
      }
    }
  }
}

// ---------------------------------------------------------------------------
// Flash attention: one block = 128 query rows of one (b,h); 8 waves x 16 rows.
// K/V tiles staged via CDNA5 async global->LDS DMA (ASYNCcnt), matmuls on
// v_wmma_f32_16x16x32_bf16, online softmax in fp32.
// ---------------------------------------------------------------------------
#define KPAD 72   // 64 bf16 + 8 pad

__global__ __launch_bounds__(256)
void flash_attn_kernel(const __bf16* __restrict__ Qws, const __bf16* __restrict__ Kws,
                       const __bf16* __restrict__ Vws, const unsigned char* __restrict__ mask,
                       const int* __restrict__ base_len, float* __restrict__ out,
                       int SQ, int SK) {
  const int H = 24, KV = 8, NREP = 3;
  __shared__ __bf16 Ks[64 * KPAD];       // K chunk  [s_local][d]
  __shared__ __bf16 Vt[64 * KPAD];       // V chunk  [d][s_local] (pre-transposed in ws)
  __shared__ __bf16 Ps[8][16 * KPAD];    // per-wave P tile (C-layout -> A-layout bridge)
  __shared__ float  maskAdd[64];

  const int tid = threadIdx.x, lane = tid & 31, wave = tid >> 5;
  const int h = blockIdx.y, b = blockIdx.z;
  const int kvh = h / NREP;
  const int qrow0 = blockIdx.x * 128 + wave * 16;

  const float scale = sqrtf(__logf((float)SQ) / __logf((float)(*base_len))) * 0.125f;

  // Q fragments live in registers for the whole key sweep
  const __bf16* Qbase = Qws + ((size_t)(b * H + h) * SQ + qrow0) * 64;
  v16bf qf[2];
#pragma unroll
  for (int c = 0; c < 2; ++c) {
    const __bf16* p = Qbase + (size_t)(lane & 15) * 64 + c * 32 + ((lane >> 4) << 3);
    v8bf lo = *(const v8bf*)p;
    v8bf hi = *(const v8bf*)(p + 16);
#pragma unroll
    for (int i = 0; i < 8; ++i) { qf[c][i] = lo[i]; qf[c][i + 8] = hi[i]; }
  }

  v8f oacc[4];
  float rm[8], rl[8];
#pragma unroll
  for (int t = 0; t < 4; ++t)
#pragma unroll
    for (int i = 0; i < 8; ++i) oacc[t][i] = 0.0f;
#pragma unroll
  for (int r = 0; r < 8; ++r) { rm[r] = -3.0e38f; rl[r] = 0.0f; }

  const __bf16* Kg = Kws + (size_t)(b * KV + kvh) * SK * 64;
  const __bf16* Vg = Vws + (size_t)(b * KV + kvh) * 64 * SK;

  // per-thread staging addresses (row/col fixed across iterations)
  const int srow = tid >> 2;                  // 0..63
  const int scb  = (tid & 3) * 16;            // 0/16/32/48
  const unsigned ldsK = (unsigned)(uintptr_t)&Ks[srow * KPAD + scb];
  const unsigned ldsV = (unsigned)(uintptr_t)&Vt[srow * KPAD + scb];

  for (int s0 = 0; s0 < SK; s0 += 64) {
    __syncthreads();   // everyone done reading previous chunk
    {
      // CDNA5 async DMA: global -> LDS without touching VGPR data path.
      unsigned long long gk =
          (unsigned long long)(uintptr_t)(Kg + (size_t)(s0 + srow) * 64 + scb);
      unsigned long long gv =
          (unsigned long long)(uintptr_t)(Vg + (size_t)srow * SK + s0 + scb);
      asm volatile(
          "global_load_async_to_lds_b128 %0, %1, off\n\t"
          "global_load_async_to_lds_b128 %0, %1, off offset:16\n\t"
          "global_load_async_to_lds_b128 %2, %3, off\n\t"
          "global_load_async_to_lds_b128 %2, %3, off offset:16"
          :
          : "v"(ldsK), "v"(gk), "v"(ldsV), "v"(gv)
          : "memory");
    }
    if (tid < 64) maskAdd[tid] = mask[(size_t)b * SK + s0 + tid] ? 0.0f : -1.0e9f;
    if (s0 + 64 < SK)
      __builtin_prefetch(Kg + (size_t)(s0 + 64 + srow) * 64, 0, 1);
    asm volatile("s_wait_asynccnt 0" ::: "memory");   // my DMAs landed in LDS
    __syncthreads();                                   // everyone's landed

    // ---- S = Q * K^T (16 x 64)
    v8f sacc[4];
#pragma unroll
    for (int nt = 0; nt < 4; ++nt) {
      v8f c;
#pragma unroll
      for (int i = 0; i < 8; ++i) c[i] = 0.0f;
      c = wmma_bf16(qf[0], frag_ld(Ks,      KPAD, nt * 16), c);   // d 0..31
      c = wmma_bf16(qf[1], frag_ld(Ks + 32, KPAD, nt * 16), c);   // d 32..63
      sacc[nt] = c;
    }

    // ---- scale + mask, per-row chunk max
    float cm[8];
#pragma unroll
    for (int r = 0; r < 8; ++r) cm[r] = -3.0e38f;
#pragma unroll
    for (int nt = 0; nt < 4; ++nt) {
      float add = maskAdd[nt * 16 + (lane & 15)];
#pragma unroll
      for (int r = 0; r < 8; ++r) {
        float x = sacc[nt][r] * scale + add;
        sacc[nt][r] = x;
        cm[r] = fmaxf(cm[r], x);
      }
    }
#pragma unroll
    for (int r = 0; r < 8; ++r)
#pragma unroll
      for (int d = 1; d < 16; d <<= 1)
        cm[r] = fmaxf(cm[r], __shfl_xor(cm[r], d, 32));

    // ---- online softmax update
    float corr[8], rs[8];
#pragma unroll
    for (int r = 0; r < 8; ++r) {
      float nm = fmaxf(rm[r], cm[r]);
      corr[r] = __expf(rm[r] - nm);
      rm[r] = nm;
      rl[r] *= corr[r];
      rs[r] = 0.0f;
    }
#pragma unroll
    for (int nt = 0; nt < 4; ++nt)
#pragma unroll
      for (int r = 0; r < 8; ++r) {
        float p = __expf(sacc[nt][r] - rm[r]);
        sacc[nt][r] = p;
        rs[r] += p;
      }
#pragma unroll
    for (int r = 0; r < 8; ++r) {
#pragma unroll
      for (int d = 1; d < 16; d <<= 1) rs[r] += __shfl_xor(rs[r], d, 32);
      rl[r] += rs[r];
    }
#pragma unroll
    for (int dt = 0; dt < 4; ++dt)
#pragma unroll
      for (int r = 0; r < 8; ++r) oacc[dt][r] *= corr[r];

    // ---- P: C-layout regs -> A-layout via per-wave LDS
    __bf16* myP = &Ps[wave][0];
    {
      int prow = ((lane >> 4) << 3), pcol = lane & 15;
#pragma unroll
      for (int nt = 0; nt < 4; ++nt)
#pragma unroll
        for (int r = 0; r < 8; ++r)
          myP[(prow + r) * KPAD + nt * 16 + pcol] = (__bf16)sacc[nt][r];
    }
    // same-wave DS ops are in order; compiler inserts s_wait_dscnt

    // ---- O += P * V
#pragma unroll
    for (int dt = 0; dt < 4; ++dt) {
      v8f c = oacc[dt];
      c = wmma_bf16(frag_ld(myP,      KPAD, 0), frag_ld(Vt,      KPAD, dt * 16), c);
      c = wmma_bf16(frag_ld(myP + 32, KPAD, 0), frag_ld(Vt + 32, KPAD, dt * 16), c);
      oacc[dt] = c;
    }
  }

  // ---- epilogue: normalize, write fp32 (B,SQ,H,D)
#pragma unroll
  for (int dt = 0; dt < 4; ++dt) {
    int d = dt * 16 + (lane & 15);
#pragma unroll
    for (int r = 0; r < 8; ++r) {
      int sr = qrow0 + r + ((lane >> 4) << 3);
      out[(((size_t)b * SQ + sr) * H + h) * 64 + d] = oacc[dt][r] / rl[r];
    }
  }
}

// ---------------------------------------------------------------------------
extern "C" void kernel_launch(void* const* d_in, const int* in_sizes, int n_in,
                              void* d_out, int out_size, void* d_ws, size_t ws_size,
                              hipStream_t stream) {
  const float* hidden = (const float*)d_in[0];
  const float* enc    = (const float*)d_in[1];
  const unsigned char* mask = (const unsigned char*)d_in[2];
  const float* qemb = (const float*)d_in[3];
  const float* kemb = (const float*)d_in[4];
  const float* wq   = (const float*)d_in[5];
  const float* wk   = (const float*)d_in[6];
  const float* wv   = (const float*)d_in[7];
  const int* base_len = (const int*)d_in[8];
  float* out = (float*)d_out;

  const int SQ = in_sizes[3] / 64;            // query_rotary_emb = SQ*32*2
  const int SK = in_sizes[4] / 64;            // key_rotary_emb   = SK*32*2
  const int B  = in_sizes[2] / SK;            // attention_mask   = B*SK
  const int QD = in_sizes[1] / (B * SK);      // encoder_hidden   = B*SK*QD

  __bf16* Qws = (__bf16*)d_ws;                       // B*24*SQ*64 bf16
  __bf16* Kws = Qws + (size_t)B * 24 * SQ * 64;      // B*8*SK*64
  __bf16* Vws = Kws + (size_t)B * 8 * SK * 64;       // B*8*64*SK (transposed)

  dim3 blk(256, 1, 1);
  proj_rope_kernel<<<dim3((24 * 64) / 128, (B * SQ) / 128, 1), blk, 0, stream>>>(
      hidden, wq, qemb, Qws, SQ, QD, 24 * 64, 0);
  proj_rope_kernel<<<dim3((8 * 64) / 128, (B * SK) / 128, 1), blk, 0, stream>>>(
      enc, wk, kemb, Kws, SK, QD, 8 * 64, 1);
  proj_rope_kernel<<<dim3((8 * 64) / 128, (B * SK) / 128, 1), blk, 0, stream>>>(
      enc, wv, nullptr, Vws, SK, QD, 8 * 64, 2);
  flash_attn_kernel<<<dim3(SQ / 128, 24, B), blk, 0, stream>>>(
      Qws, Kws, Vws, mask, base_len, out, SQ, SK);
}
